// GraphAttentionLayer_27565100106337
// MI455X (gfx1250) — matrix operand order repaired and checked by reading, not compiled, not run on hardware
//
#include <hip/hip_runtime.h>
#include <hip/hip_bf16.h>

#define N_NODES 8192
#define IN_F    512
#define OUT_F   64
#define ALPHA   0.2f
#define NEG_INF -9000000000000000.0f

typedef __attribute__((ext_vector_type(16))) _Float16     v16h;
typedef __attribute__((ext_vector_type(8)))  float        v8f;
typedef __attribute__((ext_vector_type(4)))  unsigned int v4u;
typedef __attribute__((ext_vector_type(8)))  int          v8i;
typedef __attribute__((ext_vector_type(4)))  int          v4i;

__device__ __forceinline__ float lrelu(float x) {
    return x >= 0.0f ? x : ALPHA * x;
}

// LDS byte offset of a __shared__ object (flat LDS-aperture addr truncates to
// the LDS offset per the gfx1250 aperture mapping).
__device__ __forceinline__ unsigned lds_off(const void* p) {
    return (unsigned)(unsigned long long)p;
}

// ---------------------------------------------------------------------------
// TDM: issue a 2D tile load (tile_w x tile_h elements of 4B) from global to
// LDS via the Tensor Data Mover. D# per CDNA5 ISA ch.8. Wave-uniform args;
// call from a single wave. Tracked by TENSORcnt.
// ---------------------------------------------------------------------------
__device__ __forceinline__ void tdm_load_2d(
    unsigned lds_byte_addr, const void* gptr,
    unsigned tile_w, unsigned tile_h, unsigned stride_elems) {
    const unsigned long long ga = (unsigned long long)gptr;
    v4u g0;
    g0[0] = 1u;                                   // count=1, user mode, no gather
    g0[1] = lds_byte_addr;                        // lds_addr
    g0[2] = (unsigned)(ga & 0xffffffffu);         // global_addr[31:0]
    g0[3] = ((unsigned)(ga >> 32) & 0x01ffffffu)  // global_addr[56:32]
            | (2u << 30);                         // type = 2 ("image")
    const unsigned tw = tile_w * 1u;              // tensor_dim0 == tile (in-bounds)
    const unsigned th = tile_h * 1u;              // tensor_dim1 == tile (in-bounds)
    v8i g1;
    g1[0] = (int)(2u << 16);                      // wg_mask=0, data_size=2 (4B)
    g1[1] = (int)((tw & 0xffffu) << 16);          // tensor_dim0[15:0]
    g1[2] = (int)(((tw >> 16) & 0xffffu) | ((th & 0xffffu) << 16));
    g1[3] = (int)(((th >> 16) & 0xffffu) | ((tile_w & 0xffffu) << 16)); // tile_dim0
    g1[4] = (int)(tile_h & 0xffffu);              // tile_dim1 (tile_dim2=0)
    g1[5] = (int)stride_elems;                    // tensor_dim0_stride[31:0]
    g1[6] = 0;                                    // stride hi / dim1_stride (2D: unused)
    g1[7] = 0;
    const v4i gz = {0, 0, 0, 0};
#if defined(__clang_major__) && (__clang_major__ >= 23)
    const v8i gz8 = {0, 0, 0, 0, 0, 0, 0, 0};
    __builtin_amdgcn_tensor_load_to_lds(g0, g1, gz, gz, gz8, 0);
#else
    __builtin_amdgcn_tensor_load_to_lds(g0, g1, gz, gz, 0);
#endif
}

// ---------------------------------------------------------------------------
// Kernel 0: w1 = W @ a[:64], w2 = W @ a[64:].  512 threads, 1 block.
// ---------------------------------------------------------------------------
__global__ __launch_bounds__(512) void gat_w12(
    const float* __restrict__ W, const float* __restrict__ a,
    float* __restrict__ w12) {
    const int k = threadIdx.x;
    float s1 = 0.0f, s2 = 0.0f;
#pragma unroll
    for (int j = 0; j < OUT_F; ++j) {
        float w = W[k * OUT_F + j];
        s1 += w * a[j];
        s2 += w * a[OUT_F + j];
    }
    w12[k]        = s1;
    w12[IN_F + k] = s2;
}

// ---------------------------------------------------------------------------
// Kernel 1: f1 = input @ w1, f2 = input @ w2 via WMMA f16 (16x16x32, f32 acc).
// One wave handles 16 rows; B = [w1 w2 0...0]. Fully unrolled K loop.
// ---------------------------------------------------------------------------
__global__ __launch_bounds__(256) void gat_gemv_wmma(
    const float* __restrict__ input, const float* __restrict__ w12,
    float* __restrict__ f1, float* __restrict__ f2) {
    const int lane = threadIdx.x & 31;
    const int wave = threadIdx.x >> 5;
    const int row0 = (blockIdx.x * 8 + wave) * 16;
    const int g = lane >> 4;      // lane group 0/1
    const int r = lane & 15;      // A-row within tile; B/C column index

    const float* arow = input + (size_t)(row0 + r) * IN_F;
    const float* wv   = (r == 0) ? w12 : (w12 + IN_F);
    const bool   bval = (r < 2);

    v8f c = {};
#pragma unroll
    for (int k0 = 0; k0 < IN_F; k0 += 32) {
        v16h A, B;
#pragma unroll
        for (int v = 0; v < 8; ++v) {
            // 16-bit A layout: VGPR v<4 -> K = 2v + 8g; v>=4 -> K = 16 + 2(v-4) + 8g
            const int ka = k0 + ((v < 4) ? (2 * v) : (16 + 2 * (v - 4))) + 8 * g;
            const float2 av = *reinterpret_cast<const float2*>(arow + ka);
            A[2 * v]     = (_Float16)av.x;
            A[2 * v + 1] = (_Float16)av.y;
            // B (32x16): lane = column, K same as A: v<4 -> 2v + 8g; v>=4 -> 16 + 2(v-4) + 8g (.. +1)
            const int kb = k0 + ((v < 4) ? (2 * v) : (16 + 2 * (v - 4))) + 8 * g;
            if (bval) {
                B[2 * v]     = (_Float16)wv[kb];
                B[2 * v + 1] = (_Float16)wv[kb + 1];
            } else {
                B[2 * v]     = (_Float16)0.0f;
                B[2 * v + 1] = (_Float16)0.0f;
            }
        }
        c = __builtin_amdgcn_wmma_f32_16x16x32_f16(
                false, A, false, B, (short)0, c, false, false);
    }
    // C layout: VGPR p, lanes 0-15 -> M=p, N=lane; lanes 16-31 -> M=8+p, N=lane-16
    if (r < 2) {
        float* dst = (r == 0) ? f1 : f2;
#pragma unroll
        for (int p = 0; p < 8; ++p)
            dst[row0 + p + 8 * g] = c[p];
    }
}

// ---------------------------------------------------------------------------
// Kernel 2: zero sig_scores (atomics accumulate; re-zero every call).
// ---------------------------------------------------------------------------
__global__ __launch_bounds__(256) void gat_zero(float* __restrict__ sig) {
    sig[blockIdx.x * 256 + threadIdx.x] = 0.0f;
}

// ---------------------------------------------------------------------------
// Kernel 3: per-row online softmax stats. One block = one row. The 32 KB adj
// row is TDM-streamed into LDS as two 16 KB chunks (double buffered); waves
// compute exp math out of LDS behind s_wait_tensorcnt + barrier.
// ---------------------------------------------------------------------------
#define RS_CHUNK 4096
__global__ __launch_bounds__(256) void gat_row_stats(
    const int* __restrict__ adj, const float* __restrict__ f1,
    const float* __restrict__ f2, float* __restrict__ mrow,
    float* __restrict__ invz) {
    __shared__ int abuf[2][RS_CHUNK];     // 32 KB
    const int row  = blockIdx.x;
    const int wv   = threadIdx.x >> 5;
    const float f1r = f1[row];
    const int* arow = adj + (size_t)row * N_NODES;

    if (wv == 0) {
        tdm_load_2d(lds_off(&abuf[0][0]), arow,            RS_CHUNK, 1, N_NODES);
        tdm_load_2d(lds_off(&abuf[1][0]), arow + RS_CHUNK, RS_CHUNK, 1, N_NODES);
        __builtin_amdgcn_s_wait_tensorcnt(1);   // chunk 0 resident
    }
    __syncthreads();

    float m = -3.0e38f, s = 0.0f;
    for (int jj = threadIdx.x; jj < RS_CHUNK; jj += 256) {
        const float v = (abuf[0][jj] > 0) ? lrelu(f1r + f2[jj]) : NEG_INF;
        if (v > m) { s = s * __expf(m - v) + 1.0f; m = v; }
        else       { s += __expf(v - m); }
    }

    if (wv == 0) __builtin_amdgcn_s_wait_tensorcnt(0);  // chunk 1 resident
    __syncthreads();

    for (int jj = threadIdx.x; jj < RS_CHUNK; jj += 256) {
        const float v = (abuf[1][jj] > 0) ? lrelu(f1r + f2[RS_CHUNK + jj]) : NEG_INF;
        if (v > m) { s = s * __expf(m - v) + 1.0f; m = v; }
        else       { s += __expf(v - m); }
    }

    __shared__ float sm[256];
    __shared__ float ss[256];
    sm[threadIdx.x] = m;
    ss[threadIdx.x] = s;
    __syncthreads();
#pragma unroll
    for (int off = 128; off > 0; off >>= 1) {
        if (threadIdx.x < off) {
            const float m1 = sm[threadIdx.x], s1 = ss[threadIdx.x];
            const float m2 = sm[threadIdx.x + off], s2 = ss[threadIdx.x + off];
            const float mm = fmaxf(m1, m2);
            sm[threadIdx.x] = mm;
            ss[threadIdx.x] = s1 * __expf(m1 - mm) + s2 * __expf(m2 - mm);
        }
        __syncthreads();
    }
    if (threadIdx.x == 0) {
        mrow[row] = sm[0];
        invz[row] = 1.0f / ss[0];
    }
}

// ---------------------------------------------------------------------------
// Kernel 4: column accumulation of softmax probabilities.
// grid (32 col-stripes x 64 row-chunks). Each block walks 128 rows of its
// 256-column stripe as 8 TDM tiles of 256x16 int32 (16 KB), ping-pong
// buffered; one f32 global atomic per column at the end.
// ---------------------------------------------------------------------------
#define CA_TILE_H 16
#define CA_TILES  8      // 8 * 16 = 128 rows per block
__global__ __launch_bounds__(256) void gat_col_acc(
    const int* __restrict__ adj, const float* __restrict__ f1,
    const float* __restrict__ f2, const float* __restrict__ mrow,
    const float* __restrict__ invz, float* __restrict__ sig) {
    __shared__ int abuf[2][CA_TILE_H * 256];   // 2 x 16 KB
    const int wv = threadIdx.x >> 5;
    const int j0 = blockIdx.x * 256;
    const int i0 = blockIdx.y * (CA_TILE_H * CA_TILES);
    const int j  = j0 + threadIdx.x;
    const float f2j = f2[j];

    if (wv == 0) {
        tdm_load_2d(lds_off(&abuf[0][0]),
                    adj + (size_t)i0 * N_NODES + j0, 256, CA_TILE_H, N_NODES);
    }

    float acc = 0.0f;
    for (int t = 0; t < CA_TILES; ++t) {
        if (wv == 0) {
            if (t < CA_TILES - 1) {
                tdm_load_2d(lds_off(&abuf[(t + 1) & 1][0]),
                            adj + (size_t)(i0 + (t + 1) * CA_TILE_H) * N_NODES + j0,
                            256, CA_TILE_H, N_NODES);
                __builtin_amdgcn_s_wait_tensorcnt(1);   // tile t resident
            } else {
                __builtin_amdgcn_s_wait_tensorcnt(0);   // last tile resident
            }
        }
        __syncthreads();
        const int ib = i0 + t * CA_TILE_H;
        const int* tile = &abuf[t & 1][0];
#pragma unroll
        for (int rr = 0; rr < CA_TILE_H; ++rr) {
            const int aij = tile[rr * 256 + threadIdx.x];
            if (aij > 0) {
                const int i = ib + rr;
                const float x = lrelu(f1[i] + f2j);
                acc += __expf(x - mrow[i]) * invz[i];
            }
        }
        __syncthreads();   // tile t&1 free for reuse by tile t+2
    }
    atomicAdd(&sig[j], acc);
}

// ---------------------------------------------------------------------------
// Kernel 5: exact ordered top-k by rank counting (ties -> lower index first,
// matching jax.lax.top_k). 8 blocks x 1024 threads; LDS-broadcast sweep.
// ---------------------------------------------------------------------------
__global__ __launch_bounds__(1024) void gat_topk(
    const float* __restrict__ sig, const int* __restrict__ kptr,
    int* __restrict__ out) {
    __shared__ float s[N_NODES];   // 32 KB
    for (int j = threadIdx.x; j < N_NODES; j += 1024)
        s[j] = sig[j];
    __syncthreads();

    const int   j  = blockIdx.x * 1024 + threadIdx.x;
    const float my = s[j];
    int cnt = 0;
#pragma unroll 8
    for (int k = 0; k < N_NODES; ++k) {
        const float v = s[k];
        cnt += (v > my) || (v == my && k < j);
    }
    if (cnt < kptr[0])
        out[cnt] = j;
}

// ---------------------------------------------------------------------------
extern "C" void kernel_launch(void* const* d_in, const int* in_sizes, int n_in,
                              void* d_out, int out_size, void* d_ws, size_t ws_size,
                              hipStream_t stream) {
    const float* input = (const float*)d_in[0];   // 8192 x 512
    const int*   adj   = (const int*)  d_in[1];   // 8192 x 8192
    const float* W     = (const float*)d_in[2];   // 512 x 64
    const float* a     = (const float*)d_in[3];   // 128 x 1
    const int*   kptr  = (const int*)  d_in[4];   // scalar = 512
    int*         out   = (int*)d_out;             // 512 indices

    float* ws   = (float*)d_ws;
    float* w12  = ws;                 // 1024
    float* f1   = w12 + 2 * IN_F;     // 8192
    float* f2   = f1 + N_NODES;       // 8192
    float* mrow = f2 + N_NODES;       // 8192
    float* invz = mrow + N_NODES;     // 8192
    float* sig  = invz + N_NODES;     // 8192

    gat_w12<<<1, 512, 0, stream>>>(W, a, w12);
    gat_gemv_wmma<<<N_NODES / (8 * 16), 256, 0, stream>>>(input, w12, f1, f2);
    gat_zero<<<N_NODES / 256, 256, 0, stream>>>(sig);
    gat_row_stats<<<N_NODES, 256, 0, stream>>>(adj, f1, f2, mrow, invz);
    gat_col_acc<<<dim3(N_NODES / 256, N_NODES / (CA_TILE_H * CA_TILES)), 256, 0, stream>>>(
        adj, f1, f2, mrow, invz, sig);
    gat_topk<<<N_NODES / 1024, 1024, 0, stream>>>(sig, kptr, out);
}
